// MultiHeadAttention_41910290874804
// MI455X (gfx1250) — matrix operand order repaired and checked by reading, not compiled, not run on hardware
//
#include <hip/hip_runtime.h>
#include <hip/hip_bf16.h>

// ---------------------------------------------------------------------------
// MI455X (gfx1250) fused multi-head attention, bf16 WMMA + f32 accumulate.
// Data movement via Tensor Data Mover (tensor_load_to_lds, TENSORcnt) with
// double-buffered LDS tiles; compute via v_wmma_f32_16x16x32_bf16.
// ---------------------------------------------------------------------------

#define BATCH 2
#define SEQ   2048
#define EMB   1024
#define HEADS 16
#define HDIM  64
#define MROWS (BATCH * SEQ)   // 4096

typedef __bf16 bf16;
typedef __attribute__((ext_vector_type(16))) __bf16 v16bf;
typedef __attribute__((ext_vector_type(8)))  __bf16 v8bf;
typedef __attribute__((ext_vector_type(8)))  float  v8f;

typedef __attribute__((ext_vector_type(4))) unsigned int u32x4;
typedef __attribute__((ext_vector_type(4))) int          i32x4;
typedef __attribute__((ext_vector_type(8))) int          i32x8;

#ifndef __has_builtin
#define __has_builtin(x) 0
#endif
#if __has_builtin(__builtin_amdgcn_tensor_load_to_lds) && \
    __has_builtin(__builtin_amdgcn_s_wait_tensorcnt)
#define HAVE_TDM 1
#else
#define HAVE_TDM 0
#endif

// f32 -> bf16 round-to-nearest-even
static __device__ __forceinline__ bf16 f2bf(float f) {
  unsigned u = __float_as_uint(f);
  u += 0x7FFFu + ((u >> 16) & 1u);
  unsigned short s = (unsigned short)(u >> 16);
  return __builtin_bit_cast(bf16, s);
}

// D(16x16,f32) = A(16x32,bf16) * B(32x16,bf16) + C
static __device__ __forceinline__ v8f wmma_bf16(v16bf a, v16bf b, v8f c) {
  return __builtin_amdgcn_wmma_f32_16x16x32_bf16(
      /*neg_a=*/false, a, /*neg_b=*/false, b,
      /*c_mod=*/(short)0, c, /*reuse_a=*/false, /*reuse_b=*/false);
}

// A-fragment (16x32 bf16, row-major source, ld in elements).
static __device__ __forceinline__ v16bf load_a_frag(const bf16* __restrict__ p0, int ld) {
  const int lane = threadIdx.x & 31;
  const int m = lane & 15, lh = lane >> 4;
  const bf16* p = p0 + (size_t)m * ld + lh * 16;
  union { v16bf v; v8bf h8[2]; } u;
  u.h8[0] = *(const v8bf*)(p);        // K = 16h .. 16h+7
  u.h8[1] = *(const v8bf*)(p + 8);    // K = 16h+8 .. 16h+15
  return u.v;
}

// B-fragment (32x16 bf16) from [N][K] row-major: lane = col n, 16 contiguous K.
static __device__ __forceinline__ v16bf load_b_frag(const bf16* __restrict__ bt, int ld) {
  const int lane = threadIdx.x & 31;
  const int n = lane & 15, lh = lane >> 4;
  return *(const v16bf*)(bt + (size_t)n * ld + lh * 16);
}

// 16-lane (half-wave) row reductions.
static __device__ __forceinline__ float rmax16(float v) {
  for (int off = 8; off; off >>= 1) v = fmaxf(v, __shfl_xor(v, off, 32));
  return v;
}
static __device__ __forceinline__ float rsum16(float v) {
  for (int off = 8; off; off >>= 1) v += __shfl_xor(v, off, 32);
  return v;
}

// Generic (shared-derived) pointer -> LDS byte offset (flat LDS aperture keeps
// the LDS offset in addr[31:0]).
static __device__ __forceinline__ unsigned lds_off(const void* p) {
  return (unsigned)(unsigned long long)p;
}

#if HAVE_TDM
// ---------------------------------------------------------------------------
// TDM 2D tile load: copies tile_d1 rows x tile_d0 elements (bf16) from a
// row-major tensor (row stride = stride0 elements) into contiguous LDS.
// D# layout per CDNA5 ISA (groups 0/1; groups 2/3 zero).
// This toolchain's builtin takes 6 args (g0, g1, g2, g3, extra, cpol).
// ---------------------------------------------------------------------------
static __device__ __forceinline__ void tdm_load_2d(unsigned lds_addr, const void* gptr,
                                                   unsigned tile_d0, unsigned tile_d1,
                                                   unsigned tensor_d0, unsigned tensor_d1,
                                                   unsigned long long stride0) {
  const unsigned long long ga = (unsigned long long)gptr;
  u32x4 g0;
  g0[0] = 1u;                                            // count=1 (user D#)
  g0[1] = lds_addr;                                      // lds_addr [63:32]
  g0[2] = (unsigned)(ga & 0xFFFFFFFFull);                // global_addr[31:0]
  g0[3] = (unsigned)((ga >> 32) & 0x1FFFFFFull) | (2u << 30);  // ga[56:32] | type=2
  i32x8 g1;
  g1[0] = (int)(1u << 16);                               // data_size=1 (2 bytes)
  g1[1] = (int)((tensor_d0 & 0xFFFFu) << 16);            // tensor_dim0[15:0]
  g1[2] = (int)(((tensor_d0 >> 16) & 0xFFFFu) |
                ((tensor_d1 & 0xFFFFu) << 16));          // td0[31:16] | td1[15:0]
  g1[3] = (int)(((tensor_d1 >> 16) & 0xFFFFu) |
                ((tile_d0 & 0xFFFFu) << 16));            // td1[31:16] | tile_dim0
  g1[4] = (int)(tile_d1 & 0xFFFFu);                      // tile_dim1 | tile_dim2=0
  g1[5] = (int)(unsigned)(stride0 & 0xFFFFFFFFull);      // dim0_stride[31:0]
  g1[6] = (int)(unsigned)((stride0 >> 32) & 0xFFFFull);  // dim0_stride[47:32]
  g1[7] = 0;                                             // dim1_stride=0 (2D)
  const i32x4 gz4 = {0, 0, 0, 0};
  const i32x8 gz8 = {0, 0, 0, 0, 0, 0, 0, 0};
  __builtin_amdgcn_tensor_load_to_lds(g0, g1, gz4, gz4, gz8, 0);
}
#endif

// ---------------------------------------------------------------------------
// f32 -> bf16 elementwise convert
// ---------------------------------------------------------------------------
__global__ __launch_bounds__(256) void cvt_f32_bf16(const float* __restrict__ in,
                                                    bf16* __restrict__ out,
                                                    size_t n) {
  size_t i = (size_t)blockIdx.x * blockDim.x + threadIdx.x;
  if (i < n) out[i] = f2bf(in[i]);
}

// ---------------------------------------------------------------------------
// W[K][N] f32  ->  Wt[N][K] bf16  (LDS tile transpose, 32x32)
// ---------------------------------------------------------------------------
__global__ __launch_bounds__(256) void transpose_cvt(const float* __restrict__ W,
                                                     bf16* __restrict__ Wt) {
  __shared__ float tile[32][33];
  const int c0 = blockIdx.x * 32, r0 = blockIdx.y * 32;
  const int tx = threadIdx.x & 31, ty = threadIdx.x >> 5;   // 32 x 8
  for (int i = ty; i < 32; i += 8)
    tile[i][tx] = W[(size_t)(r0 + i) * EMB + (c0 + tx)];
  __syncthreads();
  for (int i = ty; i < 32; i += 8)
    Wt[(size_t)(c0 + i) * EMB + (r0 + tx)] = f2bf(tile[tx][i]);
}

// ---------------------------------------------------------------------------
// WMMA GEMM: C[4096,1024] = A[4096,1024] x Bt[1024,1024]^T  (bf16, f32 acc)
// 64x64 block tile, 4 waves; A/B 64x64 tiles TDM'd into double-buffered LDS.
// MODE 0: Q -> bf16 [B,H,S,D] scaled 1/sqrt(D);  MODE 1: K -> bf16 [B,H,S,D]
// MODE 2: V -> bf16 [B,H,D,S];                   MODE 3: f32 + bias row-major
// ---------------------------------------------------------------------------
template <int MODE>
__global__ __launch_bounds__(128) void gemm16(const bf16* __restrict__ A,
                                              const bf16* __restrict__ Bt,
                                              void* __restrict__ Out,
                                              const float* __restrict__ bias,
                                              float scale) {
  __shared__ bf16 Asm[2][64][64];
  __shared__ bf16 Bsm[2][64][64];
  const int K = EMB, N = EMB;
  const int wave = threadIdx.x >> 5, lane = threadIdx.x & 31;
  const int row0 = blockIdx.x * 64;
  const int col0 = blockIdx.y * 64;
  const bf16* Ablk = A + (size_t)row0 * K;
  const bf16* Bblk = Bt + (size_t)col0 * K;
  v8f acc[4] = {};
  const int NKT = K / 64;   // 16 staged k-tiles

#if HAVE_TDM
  if (threadIdx.x < 32) {   // wave 0 drives the TDM (EXEC ignored by tensor ops)
    tdm_load_2d(lds_off(&Asm[0][0][0]), Ablk, 64, 64, K, 64, K);
    tdm_load_2d(lds_off(&Bsm[0][0][0]), Bblk, 64, 64, K, 64, K);
  }
#endif
  for (int kt = 0; kt < NKT; ++kt) {
    const int cur = kt & 1;
#if HAVE_TDM
    if (threadIdx.x < 32) {
      if (kt + 1 < NKT) {     // prefetch next tiles, then wait for current pair
        tdm_load_2d(lds_off(&Asm[cur ^ 1][0][0]), Ablk + (kt + 1) * 64, 64, 64, K, 64, K);
        tdm_load_2d(lds_off(&Bsm[cur ^ 1][0][0]), Bblk + (kt + 1) * 64, 64, 64, K, 64, K);
        __builtin_amdgcn_s_wait_tensorcnt(2);
      } else {
        __builtin_amdgcn_s_wait_tensorcnt(0);
      }
    }
#else
    for (int idx = threadIdx.x; idx < 64 * 8; idx += 128) {
      const int r = idx >> 3, c = (idx & 7) * 8;
      *(v8bf*)&Asm[cur][r][c] = *(const v8bf*)(Ablk + (size_t)r * K + kt * 64 + c);
      *(v8bf*)&Bsm[cur][r][c] = *(const v8bf*)(Bblk + (size_t)r * K + kt * 64 + c);
    }
#endif
    __syncthreads();
#pragma unroll
    for (int ks = 0; ks < 64; ks += 32) {
      v16bf a = load_a_frag(&Asm[cur][wave * 16][ks], 64);
#pragma unroll
      for (int j = 0; j < 4; ++j) {
        v16bf b = load_b_frag(&Bsm[cur][j * 16][ks], 64);
        acc[j] = wmma_bf16(a, b, acc[j]);
      }
    }
    __syncthreads();
  }

  // Epilogue. C layout: lane l holds col n=l&15, rows m = 8*(l>>4) + r.
  const int n = lane & 15, mb = (lane >> 4) * 8;
  const int wrow0 = row0 + wave * 16;
#pragma unroll
  for (int j = 0; j < 4; ++j) {
    const int col = col0 + j * 16 + n;
#pragma unroll
    for (int r = 0; r < 8; ++r) {
      const int i = wrow0 + mb + r;
      const float v = acc[j][r] * scale;
      if (MODE == 3) {
        ((float*)Out)[(size_t)i * N + col] = v + bias[col];
      } else {
        const int bb = i / SEQ, ss = i % SEQ;
        const int hh = col >> 6, dd = col & 63;
        const size_t idx =
            (MODE == 2) ? ((size_t)(bb * HEADS + hh) * HDIM + dd) * SEQ + ss
                        : ((size_t)(bb * HEADS + hh) * SEQ + ss) * HDIM + dd;
        ((bf16*)Out)[idx] = f2bf(v);
      }
    }
  }
}

// ---------------------------------------------------------------------------
// Flash attention (causal, online softmax).
// Grid (S/64, H, B), 128 threads = 4 waves; wave w owns queries q0..q0+15.
// Block-uniform key loop so all 4 waves share TDM-staged K (32x64) and
// Vt (64x32) tiles in double-buffered LDS. Waves past their causal limit
// only participate in barriers. P relaid out C->A via per-wave LDS tile.
// ---------------------------------------------------------------------------
__global__ __launch_bounds__(128) void flash_attn(const bf16* __restrict__ Q,
                                                  const bf16* __restrict__ Kmat,
                                                  const bf16* __restrict__ Vt,
                                                  bf16* __restrict__ ctx) {
  __shared__ bf16 Ksm[2][32][64];      // [key][d]
  __shared__ bf16 Vsm[2][64][32];      // [d][key]
  __shared__ bf16 pshare[4][16][32];   // per-wave P tile (16 q x 32 k)
  const int wave = threadIdx.x >> 5, lane = threadIdx.x & 31;
  const int b = blockIdx.z, headIdx = blockIdx.y;
  const int q0 = blockIdx.x * 64 + wave * 16;
  const int n = lane & 15, lh = lane >> 4, mb = lh * 8;

  const size_t bh = (size_t)b * HEADS + headIdx;
  const bf16* Qb = Q    + bh * SEQ * HDIM;
  const bf16* Kb = Kmat + bh * SEQ * HDIM;
  const bf16* Vb = Vt   + bh * HDIM * SEQ;

  // Q A-fragments for d-slices [0,32) and [32,64): reused for every key chunk.
  const v16bf qa0 = load_a_frag(Qb + (size_t)q0 * HDIM,      HDIM);
  const v16bf qa1 = load_a_frag(Qb + (size_t)q0 * HDIM + 32, HDIM);

  v8f o[4] = {};                 // 16 q x 64 d accumulator (4 C-frags)
  float mrow[8], lrow[8];
#pragma unroll
  for (int r = 0; r < 8; ++r) { mrow[r] = -3.0e38f; lrow[r] = 0.f; }

  const int kmax_w   = q0 + 16;              // this wave's causal key limit
  const int kmax_blk = blockIdx.x * 64 + 64; // block max (uniform loop bound)
  const int nkt = kmax_blk / 32;

#if HAVE_TDM
  if (threadIdx.x < 32) {
    tdm_load_2d(lds_off(&Ksm[0][0][0]), Kb, 64, 32, 64, 32, 64);
    tdm_load_2d(lds_off(&Vsm[0][0][0]), Vb, 32, 64, SEQ, 64, SEQ);
  }
#endif
  for (int kt = 0; kt < nkt; ++kt) {
    const int kc = kt * 32, cur = kt & 1;
#if HAVE_TDM
    if (threadIdx.x < 32) {
      if (kt + 1 < nkt) {
        const int kn = kc + 32;
        tdm_load_2d(lds_off(&Ksm[cur ^ 1][0][0]), Kb + (size_t)kn * HDIM, 64, 32, 64, 32, 64);
        tdm_load_2d(lds_off(&Vsm[cur ^ 1][0][0]), Vb + kn, 32, 64, SEQ, 64, SEQ);
        __builtin_amdgcn_s_wait_tensorcnt(2);
      } else {
        __builtin_amdgcn_s_wait_tensorcnt(0);
      }
    }
#else
    for (int idx = threadIdx.x; idx < 32 * 8; idx += 128) {   // K tile 32x64
      const int r = idx >> 3, c = (idx & 7) * 8;
      *(v8bf*)&Ksm[cur][r][c] = *(const v8bf*)(Kb + (size_t)(kc + r) * HDIM + c);
    }
    for (int idx = threadIdx.x; idx < 64 * 4; idx += 128) {   // V tile 64x32
      const int r = idx >> 2, c = (idx & 3) * 8;
      *(v8bf*)&Vsm[cur][r][c] = *(const v8bf*)(Vb + (size_t)r * SEQ + kc + c);
    }
#endif
    __syncthreads();

    if (kc < kmax_w) {
      const int ng = (kmax_w - kc >= 32) ? 2 : 1;   // valid 16-key groups

      // ---- scores: S = Q (16x64) . K^T for 32 keys (B-frags from LDS) ----
      v8f s0 = {}, s1 = {};
      {
        const bf16* kp0 = &Ksm[cur][n][lh * 16];
        s0 = wmma_bf16(qa0, *(const v16bf*)(kp0),      s0);
        s0 = wmma_bf16(qa1, *(const v16bf*)(kp0 + 32), s0);
        if (ng == 2) {
          const bf16* kp1 = &Ksm[cur][16 + n][lh * 16];
          s1 = wmma_bf16(qa0, *(const v16bf*)(kp1),      s1);
          s1 = wmma_bf16(qa1, *(const v16bf*)(kp1 + 32), s1);
        }
      }

      // ---- causal mask + online softmax ----
      float p0[8], p1[8];
#pragma unroll
      for (int r = 0; r < 8; ++r) {
        const int qi = q0 + mb + r;
        if (kc + n > qi)                  s0[r] = -3.0e38f;
        if (ng == 2 && kc + 16 + n > qi)  s1[r] = -3.0e38f;
        float vmax = (ng == 2) ? fmaxf(s0[r], s1[r]) : s0[r];
        vmax = rmax16(vmax);
        const float mnew = fmaxf(mrow[r], vmax);
        const float corr = __expf(mrow[r] - mnew);
        mrow[r] = mnew;
        p0[r] = __expf(s0[r] - mnew);
        p1[r] = (ng == 2) ? __expf(s1[r] - mnew) : 0.f;
        lrow[r] = lrow[r] * corr + rsum16(p0[r] + p1[r]);
        o[0][r] *= corr; o[1][r] *= corr; o[2][r] *= corr; o[3][r] *= corr;
      }

      // ---- C-layout P -> LDS -> A-fragment layout ----
      bf16 (*pt)[32] = pshare[wave];
#pragma unroll
      for (int r = 0; r < 8; ++r) {
        pt[mb + r][n]      = f2bf(p0[r]);
        pt[mb + r][16 + n] = f2bf(p1[r]);
      }
      asm volatile("s_wait_dscnt 0" ::: "memory");   // same-wave write->read
      const v16bf pa = load_a_frag(&pt[0][0], 32);

      // ---- O += P (16x32) . V (32x64); B-frags from LDS Vt tile ----
#pragma unroll
      for (int dg = 0; dg < 4; ++dg) {
        const bf16* vp = &Vsm[cur][dg * 16 + n][lh * 16];
        o[dg] = wmma_bf16(pa, *(const v16bf*)vp, o[dg]);
      }
    }
    __syncthreads();
  }

  // ---- normalize and write ctx[b, q, h*64 + d] (bf16 for final GEMM) ----
  bf16* cb = ctx + (size_t)b * SEQ * EMB + (size_t)headIdx * HDIM;
#pragma unroll
  for (int r = 0; r < 8; ++r) {
    const float inv = 1.0f / lrow[r];
    const size_t rowoff = (size_t)(q0 + mb + r) * EMB;
#pragma unroll
    for (int dg = 0; dg < 4; ++dg)
      cb[rowoff + dg * 16 + n] = f2bf(o[dg][r] * inv);
  }
}

// ---------------------------------------------------------------------------
// Host-side orchestration
// ---------------------------------------------------------------------------
extern "C" void kernel_launch(void* const* d_in, const int* in_sizes, int n_in,
                              void* d_out, int out_size, void* d_ws, size_t ws_size,
                              hipStream_t stream) {
  const float* emb = (const float*)d_in[0];
  const float* wq  = (const float*)d_in[1];
  const float* wk  = (const float*)d_in[2];
  const float* wv  = (const float*)d_in[3];
  const float* wo  = (const float*)d_in[4];
  const float* bo  = (const float*)d_in[5];
  float* out = (float*)d_out;

  // Workspace carve-out (~48 MB total).
  char* p = (char*)d_ws;
  auto take = [&](size_t bytes) -> void* {
    void* r = (void*)p;
    p += (bytes + 255) & ~(size_t)255;
    return r;
  };
  bf16* Xb   = (bf16*)take((size_t)MROWS * EMB * sizeof(bf16));
  bf16* WqT  = (bf16*)take((size_t)EMB * EMB * sizeof(bf16));
  bf16* WkT  = (bf16*)take((size_t)EMB * EMB * sizeof(bf16));
  bf16* WvT  = (bf16*)take((size_t)EMB * EMB * sizeof(bf16));
  bf16* WoT  = (bf16*)take((size_t)EMB * EMB * sizeof(bf16));
  bf16* Qm   = (bf16*)take((size_t)MROWS * EMB * sizeof(bf16)); // [B,H,S,D]
  bf16* Km   = (bf16*)take((size_t)MROWS * EMB * sizeof(bf16)); // [B,H,S,D]
  bf16* Vtm  = (bf16*)take((size_t)MROWS * EMB * sizeof(bf16)); // [B,H,D,S]
  bf16* Ctx  = (bf16*)take((size_t)MROWS * EMB * sizeof(bf16)); // [B,S,E]
  (void)ws_size; (void)in_sizes; (void)n_in; (void)out_size;

  // 1) Converts / transposes.
  {
    const size_t nElem = (size_t)MROWS * EMB;
    cvt_f32_bf16<<<dim3((unsigned)(nElem / 256)), 256, 0, stream>>>(emb, Xb, nElem);
  }
  const dim3 tgrid(EMB / 32, EMB / 32);
  transpose_cvt<<<tgrid, 256, 0, stream>>>(wq, WqT);
  transpose_cvt<<<tgrid, 256, 0, stream>>>(wk, WkT);
  transpose_cvt<<<tgrid, 256, 0, stream>>>(wv, WvT);
  transpose_cvt<<<tgrid, 256, 0, stream>>>(wo, WoT);

  // 2) Q/K/V projection GEMMs (Q pre-scaled by 1/sqrt(D)).
  const dim3 ggrid(MROWS / 64, EMB / 64);
  gemm16<0><<<ggrid, 128, 0, stream>>>(Xb, WqT, (void*)Qm,  nullptr, 0.125f);
  gemm16<1><<<ggrid, 128, 0, stream>>>(Xb, WkT, (void*)Km,  nullptr, 1.0f);
  gemm16<2><<<ggrid, 128, 0, stream>>>(Xb, WvT, (void*)Vtm, nullptr, 1.0f);

  // 3) Causal flash attention.
  flash_attn<<<dim3(SEQ / 64, HEADS, BATCH), 128, 0, stream>>>(Qm, Km, Vtm, Ctx);

  // 4) Output projection with bias (f32 result).
  gemm16<3><<<ggrid, 128, 0, stream>>>(Ctx, WoT, (void*)out, bo, 1.0f);
}
